// PointNet2FeatureExtractor_67929202753733
// MI455X (gfx1250) — matrix operand order repaired and checked by reading, not compiled, not run on hardware
//
#include <hip/hip_runtime.h>
#include <hip/hip_bf16.h>

typedef __attribute__((ext_vector_type(16))) _Float16 v16h;
typedef __attribute__((ext_vector_type(8)))  _Float16 v8h;
typedef __attribute__((ext_vector_type(8)))  float    v8f;

#define BB 16
#define NPT 2048

// ---------------- geometry / prep kernels ----------------

__global__ void k_xyz_transpose(const float* __restrict__ xin, float* __restrict__ xyz0, int n) {
  long t = (long)blockIdx.x * blockDim.x + threadIdx.x;
  long tot = (long)BB * 3 * n;
  if (t >= tot) return;
  int b = (int)(t / (3 * n));
  int r = (int)((t / n) % 3);
  int p = (int)(t % n);
  xyz0[((long)b * n + p) * 3 + r] = xin[t];
}

__global__ void k_f32_to_f16(const float* __restrict__ in, _Float16* __restrict__ out, long tot) {
  long t = (long)blockIdx.x * blockDim.x + threadIdx.x;
  if (t >= tot) return;
  out[t] = (_Float16)in[t];
}

// Farthest point sampling: one block per batch, dist kept in LDS.
__global__ void k_fps(const float* __restrict__ xyz, int n, int npoint, int* __restrict__ idx) {
  __shared__ float sdist[2048];
  __shared__ float sval[256];
  __shared__ int   sidx[256];
  __shared__ int   sfar;
  int b = blockIdx.x, tid = threadIdx.x;
  const float* p = xyz + (long)b * n * 3;
  int* out = idx + (long)b * npoint;
  for (int i = tid; i < n; i += 256) sdist[i] = 1e10f;
  __syncthreads();
  int far = 0;
  for (int s = 0; s < npoint; s++) {
    if (tid == 0) out[s] = far;
    float fx = p[far * 3 + 0], fy = p[far * 3 + 1], fz = p[far * 3 + 2];
    float best = -1.0f; int bi = 0x7fffffff;
    for (int i = tid; i < n; i += 256) {
      float dx = p[i * 3 + 0] - fx, dy = p[i * 3 + 1] - fy, dz = p[i * 3 + 2] - fz;
      float dd = dx * dx + dy * dy + dz * dz;
      float nv = fminf(sdist[i], dd);
      sdist[i] = nv;
      if (nv > best) { best = nv; bi = i; }
    }
    sval[tid] = best; sidx[tid] = bi;
    __syncthreads();
    for (int o = 128; o > 0; o >>= 1) {
      if (tid < o) {
        float v2 = sval[tid + o]; int i2 = sidx[tid + o];
        if (v2 > sval[tid] || (v2 == sval[tid] && i2 < sidx[tid])) { sval[tid] = v2; sidx[tid] = i2; }
      }
      __syncthreads();
    }
    if (tid == 0) sfar = sidx[0];
    __syncthreads();
    far = sfar;
  }
}

__global__ void k_gather_xyz(const float* __restrict__ xyz, const int* __restrict__ idx,
                             int n, int np, float* __restrict__ nxyz) {
  int t = blockIdx.x * blockDim.x + threadIdx.x;
  if (t >= BB * np) return;
  int b = t / np;
  int g = idx[t];
  nxyz[(long)t * 3 + 0] = xyz[((long)b * n + g) * 3 + 0];
  nxyz[(long)t * 3 + 1] = xyz[((long)b * n + g) * 3 + 1];
  nxyz[(long)t * 3 + 2] = xyz[((long)b * n + g) * 3 + 2];
}

// Ball query: first 16 in-radius indices in ascending order (matches sort+take semantics).
__global__ void k_ball(const float* __restrict__ xyz, const float* __restrict__ nxyz,
                       int n, int np, float r2, int* __restrict__ gidx) {
  int t = blockIdx.x * blockDim.x + threadIdx.x;
  if (t >= BB * np) return;
  int b = t / np;
  const float* p = xyz + (long)b * n * 3;
  float qx = nxyz[(long)t * 3 + 0], qy = nxyz[(long)t * 3 + 1], qz = nxyz[(long)t * 3 + 2];
  int buf[16]; int cnt = 0;
  for (int j = 0; j < n && cnt < 16; j++) {
    float dx = p[j * 3 + 0] - qx, dy = p[j * 3 + 1] - qy, dz = p[j * 3 + 2] - qz;
    if (dx * dx + dy * dy + dz * dz <= r2) buf[cnt++] = j;
  }
  int first = (cnt > 0) ? buf[0] : 0;
  for (int k = cnt; k < 16; k++) buf[k] = first;
  int* o = gidx + (long)t * 16;
  for (int k = 0; k < 16; k++) o[k] = buf[k];
}

// Build grouped/concat input: [gxyz - new_xyz (3), gathered feat (featC), zero pad] as f16, K padded.
__global__ void k_sa_group(const float* __restrict__ xyz, const float* __restrict__ nxyz,
                           const int* __restrict__ gidx, const _Float16* __restrict__ feat,
                           int featC, int n, int np, int Kp, _Float16* __restrict__ gin) {
  long t = (long)blockIdx.x * blockDim.x + threadIdx.x;
  long tot = (long)BB * np * 16 * Kp;
  if (t >= tot) return;
  long row = t / Kp;
  int col = (int)(t % Kp);
  long bs = row >> 4;                // b*np + s
  int b = (int)(bs / np);
  int g = gidx[row];
  _Float16 v;
  if (col < 3) {
    v = (_Float16)(xyz[((long)b * n + g) * 3 + col] - nxyz[bs * 3 + col]);
  } else if (col < 3 + featC) {
    v = feat[((long)b * n + g) * featC + (col - 3)];
  } else {
    v = (_Float16)0.f;
  }
  gin[t] = v;
}

// f32 weights (outC x inC) -> f16 (outC x Kp) zero-padded.
__global__ void k_wpad(const float* __restrict__ w, _Float16* __restrict__ wp,
                       int inC, int Kp, int outC) {
  int t = blockIdx.x * blockDim.x + threadIdx.x;
  if (t >= outC * Kp) return;
  int o = t / Kp, k = t % Kp;
  wp[t] = (k < inC) ? (_Float16)w[(long)o * inC + k] : (_Float16)0.f;
}

// ---------------- WMMA GEMM: H[M,N] = A[M,K] * W[N,K]^T + bias; atomic BN stats ----------------
// Block: 256 threads (8 waves), 128x64 tile. Wave (wm 0..3, wn 0..1) owns a 32x32 output
// (4 accumulators -> 4 WMMAs per K-step). LDS tiles double-buffered, pre-swizzled into exact
// WMMA fragment order:
//   A 16x32 f16 fragment: lane = (m&15) + 16*((k%16)>=8), e = (k%8) + 8*(k>=16)
//   B 32x16 f16 fragment: lane = (n&15) + 16*((k%16)>=8), e = (k%8) + 8*(k>=16)
// Each thread's 8-half (16B) global load lands as one contiguous 16B LDS store.
// One barrier per K-step: the post-store barrier of step i-1 protects the buffer
// being overwritten at step i (reads of that buffer finished before it).
__global__ void k_wmma_gemm(const _Float16* __restrict__ A, const _Float16* __restrict__ W,
                            const float* __restrict__ bias, _Float16* __restrict__ H,
                            float* __restrict__ stats, int M, int N, int K) {
  __shared__ __align__(32) _Float16 lA[2 * 8 * 512];   // 2 bufs x 8 subtiles x (32 lanes x 16 halfs)
  __shared__ __align__(32) _Float16 lB[2 * 4 * 512];   // 2 bufs x 4 subtiles
  int tid = threadIdx.x, lane = tid & 31, wave = tid >> 5;
  int tilesN = N >> 6;
  long bm = blockIdx.x / tilesN;
  int  bn = blockIdx.x % tilesN;
  long row0 = bm << 7;           // 128-row tile
  int  col0 = bn << 6;           // 64-col tile

  // Loader: thread covers A rows (tid>>2) and (tid>>2)+64, chunk lc = tid&3 (k = 8*lc..8*lc+7),
  // and B row (tid>>2), chunk lc.
  int lr = tid >> 2;             // 0..63
  int lc = tid & 3;
  int aOff = ((((lr >> 4) * 32) + (lr & 15) + ((lc & 1) << 4)) << 4) + ((lc >> 1) << 3);
  int bOff = ((((lr >> 4) * 32) + (lr & 15) + ((lc & 1) << 4)) << 4) + ((lc >> 1) << 3);
  const _Float16* aSrc0 = A + (row0 + lr) * (long)K + (lc << 3);
  const _Float16* aSrc1 = aSrc0 + (long)64 * K;
  const _Float16* wSrc  = W + ((long)(col0 + lr)) * K + (lc << 3);

  int wm = wave >> 1, wn = wave & 1;
  int fa0 = ((2 * wm) * 32 + lane) << 4;
  int fa1 = ((2 * wm + 1) * 32 + lane) << 4;
  int fb0 = ((2 * wn) * 32 + lane) << 4;
  int fb1 = ((2 * wn + 1) * 32 + lane) << 4;

  v8f acc00 = {}, acc01 = {}, acc10 = {}, acc11 = {};

  int nSteps = K >> 5;
  // Prologue: stage step 0 into buffer 0.
  {
    v8h a0 = *(const v8h*)(aSrc0);
    v8h a1 = *(const v8h*)(aSrc1);
    v8h wv = *(const v8h*)(wSrc);
    *(v8h*)(lA + aOff) = a0;
    *(v8h*)(lA + aOff + 2048) = a1;     // rows +64 => subtile +4 => +4*512 halfs
    *(v8h*)(lB + bOff) = wv;
  }
  __syncthreads();

  int buf = 0;
  for (int i = 0; i < nSteps; i++) {
    bool more = (i + 1 < nSteps);
    v8h na0, na1, nwv;                  // only touched under uniform `more`
    if (more) {
      int kn = (i + 1) << 5;
      na0 = *(const v8h*)(aSrc0 + kn);
      na1 = *(const v8h*)(aSrc1 + kn);
      nwv = *(const v8h*)(wSrc + kn);
      if (i + 2 < nSteps) {
        int kp = (i + 2) << 5;
        __builtin_prefetch((const void*)(aSrc0 + kp), 0, 3);
        __builtin_prefetch((const void*)(aSrc1 + kp), 0, 3);
        __builtin_prefetch((const void*)(wSrc + kp), 0, 3);
      }
    }
    const _Float16* pA = lA + buf * 4096;
    const _Float16* pB = lB + buf * 2048;
    v16h a0 = *(const v16h*)(pA + fa0);
    v16h a1 = *(const v16h*)(pA + fa1);
    v16h b0 = *(const v16h*)(pB + fb0);
    v16h b1 = *(const v16h*)(pB + fb1);
    acc00 = __builtin_amdgcn_wmma_f32_16x16x32_f16(false, a0, false, b0, (short)0, acc00, false, false);
    acc01 = __builtin_amdgcn_wmma_f32_16x16x32_f16(false, a0, false, b1, (short)0, acc01, false, false);
    acc10 = __builtin_amdgcn_wmma_f32_16x16x32_f16(false, a1, false, b0, (short)0, acc10, false, false);
    acc11 = __builtin_amdgcn_wmma_f32_16x16x32_f16(false, a1, false, b1, (short)0, acc11, false, false);
    if (more) {
      int nb = buf ^ 1;
      _Float16* dA = lA + nb * 4096;
      _Float16* dB = lB + nb * 2048;
      *(v8h*)(dA + aOff) = na0;
      *(v8h*)(dA + aOff + 2048) = na1;
      *(v8h*)(dB + bOff) = nwv;
      __syncthreads();
      buf = nb;
    }
  }

  // C/D layout: lane L col = L%16; VGPR r row = r + 8*(L>=16).
  int colA = col0 + (wn << 5) + (lane & 15);
  int colB = colA + 16;
  long r0 = row0 + (wm << 5) + ((lane >> 4) << 3);
  float ba = bias[colA], bb2 = bias[colB];
  float sA = 0.f, qA = 0.f, sB = 0.f, qB = 0.f;
  for (int r = 0; r < 8; r++) {
    long ra = r0 + r, rb = r0 + 16 + r;
    float x0 = acc00[r] + ba;
    float y0 = acc01[r] + bb2;
    float x1 = acc10[r] + ba;
    float y1 = acc11[r] + bb2;
    H[ra * N + colA] = (_Float16)x0;
    H[ra * N + colB] = (_Float16)y0;
    H[rb * N + colA] = (_Float16)x1;
    H[rb * N + colB] = (_Float16)y1;
    sA += x0 + x1; qA += x0 * x0 + x1 * x1;
    sB += y0 + y1; qB += y0 * y0 + y1 * y1;
  }
  atomicAdd(&stats[colA], sA);
  atomicAdd(&stats[N + colA], qA);
  atomicAdd(&stats[colB], sB);
  atomicAdd(&stats[N + colB], qB);
}

__global__ void k_bn_finalize(float* __restrict__ stats, const float* __restrict__ gamma,
                              const float* __restrict__ beta, float invM, int N) {
  int c = blockIdx.x * blockDim.x + threadIdx.x;
  if (c >= N) return;
  float mu = stats[c] * invM;
  float var = fmaxf(stats[N + c] * invM - mu * mu, 0.f);
  float a = gamma[c] * rsqrtf(var + 1e-5f);
  stats[2 * N + c] = a;
  stats[3 * N + c] = beta[c] - mu * a;
}

__global__ void k_bn_relu(_Float16* __restrict__ H, const float* __restrict__ stats,
                          int N, long tot) {
  long t = (long)blockIdx.x * blockDim.x + threadIdx.x;
  if (t >= tot) return;
  int c = (int)(t % N);
  float x = (float)H[t] * stats[2 * N + c] + stats[3 * N + c];
  H[t] = (_Float16)fmaxf(x, 0.f);
}

__global__ void k_maxpool(const _Float16* __restrict__ H, int C, long rows,
                          _Float16* __restrict__ out) {
  long t = (long)blockIdx.x * blockDim.x + threadIdx.x;
  if (t >= rows * C) return;
  long s = t / C;
  int c = (int)(t % C);
  const _Float16* base = H + (s * 16) * (long)C + c;
  float m = -1e30f;
  for (int k = 0; k < 16; k++) m = fmaxf(m, (float)base[(long)k * C]);
  out[t] = (_Float16)m;
}

// ---------------- FP (feature propagation) kernels ----------------

__global__ void k_three_nn(const float* __restrict__ xyz1, const float* __restrict__ xyz2,
                           int n1, int n2, int* __restrict__ idx3, float* __restrict__ w3) {
  int t = blockIdx.x * blockDim.x + threadIdx.x;
  if (t >= BB * n1) return;
  int b = t / n1;
  const float* p2 = xyz2 + (long)b * n2 * 3;
  float qx = xyz1[(long)t * 3 + 0], qy = xyz1[(long)t * 3 + 1], qz = xyz1[(long)t * 3 + 2];
  float d0 = 1e30f, d1 = 1e30f, d2 = 1e30f;
  int i0 = 0, i1 = 0, i2 = 0;
  for (int j = 0; j < n2; j++) {
    float dx = p2[j * 3 + 0] - qx, dy = p2[j * 3 + 1] - qy, dz = p2[j * 3 + 2] - qz;
    float dd = dx * dx + dy * dy + dz * dz;
    if (dd < d0)      { d2 = d1; i2 = i1; d1 = d0; i1 = i0; d0 = dd; i0 = j; }
    else if (dd < d1) { d2 = d1; i2 = i1; d1 = dd; i1 = j; }
    else if (dd < d2) { d2 = dd; i2 = j; }
  }
  float w0 = 1.f / (fmaxf(d0, 0.f) + 1e-8f);
  float w1 = 1.f / (fmaxf(d1, 0.f) + 1e-8f);
  float w2 = 1.f / (fmaxf(d2, 0.f) + 1e-8f);
  float ws = w0 + w1 + w2;
  idx3[(long)t * 3 + 0] = i0; idx3[(long)t * 3 + 1] = i1; idx3[(long)t * 3 + 2] = i2;
  w3[(long)t * 3 + 0] = w0 / ws; w3[(long)t * 3 + 1] = w1 / ws; w3[(long)t * 3 + 2] = w2 / ws;
}

__global__ void k_fp_concat(const _Float16* __restrict__ p1, int C1,
                            const _Float16* __restrict__ p2, int C2,
                            const int* __restrict__ idx3, const float* __restrict__ w3,
                            int n1, int n2, int Kp, _Float16* __restrict__ gin) {
  long t = (long)blockIdx.x * blockDim.x + threadIdx.x;
  long tot = (long)BB * n1 * Kp;
  if (t >= tot) return;
  long row = t / Kp;
  int col = (int)(t % Kp);
  _Float16 v;
  if (col < C1) {
    v = p1[row * C1 + col];
  } else if (col < C1 + C2) {
    int b = (int)(row / n1);
    int c = col - C1;
    const _Float16* base = p2 + ((long)b * n2) * C2 + c;
    const int* ii = idx3 + row * 3;
    const float* ww = w3 + row * 3;
    float s = ww[0] * (float)base[(long)ii[0] * C2]
            + ww[1] * (float)base[(long)ii[1] * C2]
            + ww[2] * (float)base[(long)ii[2] * C2];
    v = (_Float16)s;
  } else {
    v = (_Float16)0.f;
  }
  gin[t] = v;
}

__global__ void k_out(const _Float16* __restrict__ act, float* __restrict__ out) {
  long t = (long)blockIdx.x * blockDim.x + threadIdx.x;
  long tot = (long)BB * 128 * NPT;
  if (t >= tot) return;
  int b = (int)(t / (128 * NPT));
  int c = (int)((t / NPT) % 128);
  int p = (int)(t % NPT);
  out[t] = (float)act[((long)b * NPT + p) * 128 + c];
}

// ---------------- host orchestration ----------------

extern "C" void kernel_launch(void* const* d_in, const int* in_sizes, int n_in,
                              void* d_out, int out_size, void* d_ws, size_t ws_size,
                              hipStream_t stream) {
  (void)in_sizes; (void)n_in; (void)out_size; (void)ws_size;

  static const int   nL[5]      = {2048, 1024, 512, 256, 128};
  static const int   sa_np[4]   = {1024, 512, 256, 128};
  static const float sa_r[4]    = {0.05f, 0.1f, 0.2f, 0.4f};
  static const int   sa_mlp[4][3] = {{512,256,256},{256,128,64},{64,128,256},{256,256,512}};

  size_t off = 0;
  auto alloc = [&](size_t bytes) -> void* {
    void* p = (char*)d_ws + off;
    off = (off + bytes + 255) & ~(size_t)255;
    return p;
  };

  float* xyzL[5];
  for (int i = 0; i < 5; i++) xyzL[i] = (float*)alloc((size_t)BB * nL[i] * 3 * 4);
  _Float16* l0f    = (_Float16*)alloc((size_t)BB * NPT * 3 * 2);
  int*      fpsidx = (int*)alloc((size_t)BB * 1024 * 4);
  int*      gidx   = (int*)alloc((size_t)BB * 1024 * 16 * 4);
  int*      idx3   = (int*)alloc((size_t)BB * NPT * 3 * 4);
  float*    w3     = (float*)alloc((size_t)BB * NPT * 3 * 4);
  float*    stats  = (float*)alloc((size_t)4 * 512 * 4);
  _Float16* wpad   = (_Float16*)alloc((size_t)256 * 768 * 2);
  _Float16* featL1 = (_Float16*)alloc((size_t)BB * 1024 * 256 * 2);
  _Float16* featL2 = (_Float16*)alloc((size_t)BB * 512 * 64 * 2);
  _Float16* featL3 = (_Float16*)alloc((size_t)BB * 256 * 256 * 2);
  _Float16* featL4 = (_Float16*)alloc((size_t)BB * 128 * 512 * 2);
  _Float16* fpo4   = (_Float16*)alloc((size_t)BB * 256 * 256 * 2);
  _Float16* fpo3   = (_Float16*)alloc((size_t)BB * 512 * 256 * 2);
  _Float16* fpo2   = (_Float16*)alloc((size_t)BB * 1024 * 128 * 2);
  _Float16* fpo1   = (_Float16*)alloc((size_t)BB * 2048 * 128 * 2);
  _Float16* GIN    = (_Float16*)alloc((size_t)131072 * 288 * 2);
  _Float16* PING   = (_Float16*)alloc((size_t)262144 * 512 * 2);
  _Float16* PONG   = (_Float16*)alloc((size_t)262144 * 256 * 2);

  auto runLayer = [&](const _Float16* Ain, _Float16* Hout, int M, int Kp,
                      int inC, int outC, int lidx) {
    const float* w  = (const float*)d_in[1 + 4 * lidx + 0];
    const float* bi = (const float*)d_in[1 + 4 * lidx + 1];
    const float* ga = (const float*)d_in[1 + 4 * lidx + 2];
    const float* be = (const float*)d_in[1 + 4 * lidx + 3];
    int wn = outC * Kp;
    k_wpad<<<(wn + 255) / 256, 256, 0, stream>>>(w, wpad, inC, Kp, outC);
    hipMemsetAsync(stats, 0, (size_t)2 * outC * sizeof(float), stream);
    dim3 grid((unsigned)((M / 128) * (outC / 64)));
    k_wmma_gemm<<<grid, 256, 0, stream>>>(Ain, wpad, bi, Hout, stats, M, outC, Kp);
    k_bn_finalize<<<(outC + 63) / 64, 64, 0, stream>>>(stats, ga, be, 1.0f / (float)M, outC);
    long tot = (long)M * outC;
    k_bn_relu<<<(unsigned)((tot + 255) / 256), 256, 0, stream>>>(Hout, stats, outC, tot);
  };

  // l0: transpose (B,3,N) -> (B,N,3), plus f16 copy for feature concat
  {
    long tot = (long)BB * 3 * NPT;
    k_xyz_transpose<<<(unsigned)((tot + 255) / 256), 256, 0, stream>>>((const float*)d_in[0], xyzL[0], NPT);
    k_f32_to_f16<<<(unsigned)((tot + 255) / 256), 256, 0, stream>>>(xyzL[0], l0f, tot);
  }

  // ---- SA levels ----
  const _Float16* featPrev = l0f;
  int featC = 3;
  _Float16* featBuf[4] = {featL1, featL2, featL3, featL4};
  int lidx = 0;
  for (int lvl = 0; lvl < 4; lvl++) {
    int n = nL[lvl], np = sa_np[lvl];
    k_fps<<<BB, 256, 0, stream>>>(xyzL[lvl], n, np, fpsidx);
    k_gather_xyz<<<(BB * np + 255) / 256, 256, 0, stream>>>(xyzL[lvl], fpsidx, n, np, xyzL[lvl + 1]);
    k_ball<<<(BB * np + 255) / 256, 256, 0, stream>>>(xyzL[lvl], xyzL[lvl + 1], n, np,
                                                      sa_r[lvl] * sa_r[lvl], gidx);
    int inC = 3 + featC;
    int Kp = ((inC + 31) / 32) * 32;
    long M = (long)BB * np * 16;
    long tot = M * Kp;
    k_sa_group<<<(unsigned)((tot + 255) / 256), 256, 0, stream>>>(xyzL[lvl], xyzL[lvl + 1], gidx,
                                                                  featPrev, featC, n, np, Kp, GIN);
    runLayer(GIN,  PING, (int)M, Kp,             inC,            sa_mlp[lvl][0], lidx++);
    runLayer(PING, PONG, (int)M, sa_mlp[lvl][0], sa_mlp[lvl][0], sa_mlp[lvl][1], lidx++);
    runLayer(PONG, PING, (int)M, sa_mlp[lvl][1], sa_mlp[lvl][1], sa_mlp[lvl][2], lidx++);
    long rows = (long)BB * np;
    k_maxpool<<<(unsigned)((rows * sa_mlp[lvl][2] + 255) / 256), 256, 0, stream>>>(
        PING, sa_mlp[lvl][2], rows, featBuf[lvl]);
    featPrev = featBuf[lvl];
    featC = sa_mlp[lvl][2];
  }

  // ---- FP levels (order fp4, fp3, fp2, fp1 to match params order) ----
  struct FPcfg {
    int lvl; const _Float16* p1; int C1; const _Float16* p2; int C2;
    _Float16* out; int m0, m1;
  };
  FPcfg fpc[4] = {
    {3, featL3, 256, featL4, 512, fpo4, 256, 256},
    {2, featL2, 64,  fpo4,   256, fpo3, 256, 256},
    {1, featL1, 256, fpo3,   256, fpo2, 256, 128},
    {0, l0f,    3,   fpo2,   128, fpo1, 128, 128},
  };
  for (int i = 0; i < 4; i++) {
    FPcfg f = fpc[i];
    int n1 = nL[f.lvl], n2 = nL[f.lvl + 1];
    k_three_nn<<<(BB * n1 + 255) / 256, 256, 0, stream>>>(xyzL[f.lvl], xyzL[f.lvl + 1],
                                                          n1, n2, idx3, w3);
    int inC = f.C1 + f.C2;
    int Kp = ((inC + 31) / 32) * 32;
    long M = (long)BB * n1;
    long tot = M * Kp;
    k_fp_concat<<<(unsigned)((tot + 255) / 256), 256, 0, stream>>>(f.p1, f.C1, f.p2, f.C2,
                                                                   idx3, w3, n1, n2, Kp, GIN);
    runLayer(GIN,  PING,  (int)M, Kp,   inC,  f.m0, lidx++);
    runLayer(PING, f.out, (int)M, f.m0, f.m0, f.m1, lidx++);
  }

  // ---- output transpose (B,N,128)->(B,128,N) f32 ----
  {
    long tot = (long)BB * 128 * NPT;
    k_out<<<(unsigned)((tot + 255) / 256), 256, 0, stream>>>(fpo1, (float*)d_out);
  }
}
